// ScoringLayer_67250597921428
// MI455X (gfx1250) — matrix operand order repaired and checked by reading, not compiled, not run on hardware
//
#include <hip/hip_runtime.h>

typedef __attribute__((ext_vector_type(16))) _Float16 v16h;
typedef __attribute__((ext_vector_type(4)))  _Float16 v4h;
typedef __attribute__((ext_vector_type(8)))  float    v8f;
typedef __attribute__((ext_vector_type(4)))  float    f4;

#define VOCAB  2048
#define DDIM   16384   // 8 positions * 2048 vocab
#define BATCH  256
#define KSTEP  64      // K per staged block: 2 x wmma_f32_16x16x32_f16
#define MROWS  128     // batch rows per workgroup (8 waves * 16)
#define NTILE  16      // output columns per workgroup
#define LOG2_EPS (-26.575424759098898f)  // log2(1e-8)
#define LOG2E    (1.4426950408889634f)

// ---------------------------------------------------------------------------
// Kernel 1: inv_n[b] = 1 / max(sum_d binary[b,d], 1)
// ---------------------------------------------------------------------------
__global__ __launch_bounds__(256) void rowsum_kernel(const float* __restrict__ binary,
                                                     float* __restrict__ inv_n) {
  __shared__ float red[256];
  const int b = blockIdx.x;
  const float* row = binary + (size_t)b * DDIM;
  float s = 0.0f;
  for (int i = threadIdx.x; i < DDIM; i += 256) s += row[i];
  red[threadIdx.x] = s;
  __syncthreads();
  for (int off = 128; off > 0; off >>= 1) {
    if ((int)threadIdx.x < off) red[threadIdx.x] += red[threadIdx.x + off];
    __syncthreads();
  }
  if (threadIdx.x == 0) inv_n[b] = 1.0f / fmaxf(red[0], 1.0f);
}

// ---------------------------------------------------------------------------
// Kernel 2: fused  out[b,v] = 2^( (binary[b,:] . log2_w[v,:]) * inv_n[b] )
// Base-2 domain (v_exp_f32/v_log_f32 are natively exp2/log2; ln2 cancels).
// A (binary) rows are private to one wave each -> loaded straight from
// global into WMMA fragment layout, no LDS round-trip.  Only the shared
// B tile (log2-weights, used by all 8 waves) is staged & double-buffered
// in LDS.
// ---------------------------------------------------------------------------
__global__ __launch_bounds__(256) void scoring_wmma_kernel(
    const float* __restrict__ binary, const float* __restrict__ raw,
    const float* __restrict__ inv_n, const int* __restrict__ pmaxw,
    float* __restrict__ out) {
  __shared__ __align__(32) _Float16 ldsB[2][NTILE][KSTEP];  // 4 KiB
  __shared__ float ldsN[MROWS];

  const int tid  = threadIdx.x;
  const int wave = tid >> 5;      // 0..7, one 16x16 M-tile per wave
  const int lane = tid & 31;
  const int lh   = lane >> 4;     // lane half (K-split in WMMA layouts)
  const int ll   = lane & 15;     // row (A) / col (B,C) within tile
  const int m_base = blockIdx.y * MROWS;
  const int n_base = blockIdx.x * NTILE;
  const float log2_max_w = __builtin_amdgcn_logf((float)pmaxw[0]);

  if (tid < MROWS) ldsN[tid] = inv_n[m_base + tid];

  // A: this lane's row, direct-from-global fragment loads.
  const int arow = wave * 16 + ll;
  const float* aptr = binary + (size_t)(m_base + arow) * DDIM + 16 * lh;

  // B staging coordinates: 16x64 tile, 4 elems/thread.
  const int bn = tid >> 4;              // B tile row (n) 0..15
  const int bk = (tid & 15) * 4;        // B tile col (k), multiple of 4
  const float* bptr = raw + (size_t)(n_base + bn) * DDIM + bk;

  auto loadB = [&](int k0, int bufi) {
    // log2(max(maxw*sigmoid(x), eps)) via raw HW exp2/log2 (no fixups).
    f4 y = *(const f4*)(bptr + k0);
    v4h lw;
#pragma unroll
    for (int i = 0; i < 4; ++i) {
      float t = __builtin_amdgcn_exp2f(-y[i] * LOG2E);          // e^-x
      float l = log2_max_w - __builtin_amdgcn_logf(1.0f + t);   // log2(maxw*sig)
      lw[i] = (_Float16)fmaxf(l, LOG2_EPS);
    }
    *(v4h*)&ldsB[bufi][bn][bk] = lw;
  };

  v8f acc = {};

  auto computeblk = [&](int k0, int bufi) {
    // A fragment layout (16-bit A 16x32): lane holds row lane&15, K
    // [16*lh, 16*lh+16) of each 32-wide sub-block (matches B).
#pragma unroll
    for (int s = 0; s < 2; ++s) {                // two K=32 sub-blocks
      const int kb = k0 + 32 * s;
      f4 a0 = *(const f4*)(aptr + kb);           // K 16lh .. 16lh+3
      f4 a1 = *(const f4*)(aptr + kb + 4);       // K 16lh+4 .. 16lh+7
      f4 a2 = *(const f4*)(aptr + kb + 8);       // K 16lh+8 ..
      f4 a3 = *(const f4*)(aptr + kb + 12);
      v16h afrag;
#pragma unroll
      for (int i = 0; i < 4; ++i) {
        afrag[i]      = (_Float16)a0[i];
        afrag[4 + i]  = (_Float16)a1[i];
        afrag[8 + i]  = (_Float16)a2[i];
        afrag[12 + i] = (_Float16)a3[i];
      }
      // B fragment (32x16): col = lane&15, K 0..15 / 16..31 across halves.
      v16h bfrag = *(const v16h*)&ldsB[bufi][ll][32 * s + 16 * lh];
      acc = __builtin_amdgcn_wmma_f32_16x16x32_f16(false, afrag, false, bfrag,
                                                   (short)0, acc, false, false);
    }
  };

  // Branch-free double-buffered pipeline over the shared B tile.
  loadB(0, 0);
  __syncthreads();
  int k0 = 0;
#pragma unroll 1
  for (; k0 + 2 * KSTEP < DDIM; k0 += 2 * KSTEP) {
    loadB(k0 + KSTEP, 1);
    computeblk(k0, 0);
    __syncthreads();
    loadB(k0 + 2 * KSTEP, 0);
    computeblk(k0 + KSTEP, 1);
    __syncthreads();
  }
  loadB(k0 + KSTEP, 1);   // last staged block
  computeblk(k0, 0);
  __syncthreads();
  computeblk(k0 + KSTEP, 1);

  // Epilogue: C VGPR r -> row r + 8*(lane>>4), col lane&15.
  // out = 2^(acc * inv_n): one v_mul + one v_exp_f32 per element.
#pragma unroll
  for (int r = 0; r < 8; ++r) {
    const int lm = wave * 16 + r + 8 * lh;
    const float v = __builtin_amdgcn_exp2f(acc[r] * ldsN[lm]);
    out[(size_t)(m_base + lm) * VOCAB + (n_base + ll)] = v;
  }
}

// ---------------------------------------------------------------------------
extern "C" void kernel_launch(void* const* d_in, const int* in_sizes, int n_in,
                              void* d_out, int out_size, void* d_ws, size_t ws_size,
                              hipStream_t stream) {
  const float* binary = (const float*)d_in[0];
  const float* raw    = (const float*)d_in[1];
  const int*   pmaxw  = (const int*)d_in[2];  // python int scalar -> int[1]
  float* inv_n = (float*)d_ws;                 // 256 floats of scratch

  rowsum_kernel<<<BATCH, 256, 0, stream>>>(binary, inv_n);

  dim3 grid(VOCAB / NTILE, BATCH / MROWS);     // (128, 2)
  scoring_wmma_kernel<<<grid, 256, 0, stream>>>(binary, raw, inv_n, pmaxw,
                                                (float*)d_out);
}
